// Net_63660005262127
// MI455X (gfx1250) — compile-verified, matrix operand order checked by reading, not run on hardware
//
#include <hip/hip_runtime.h>
#include <hip/hip_bf16.h>
#include <stdint.h>

typedef __attribute__((ext_vector_type(16))) _Float16 v16h;
typedef __attribute__((ext_vector_type(8)))  float    v8f;

// =================== LDS layout (149760 B total) ===================
// [0, 112896): f32 NHWC frame (async DMA target), later reused:
#define OFF_G1    0        // f16 [16][27][32] = 27648   (gaze26 out, pad row+cols)
#define OFF_A2    27648    // f16 [16][12][16] = 6144    (conv2 out, pad row)
#define OFF_W2S   33792    // f16 [20][2][16][16] = 20480 (conv2 B, swizzled, staged late)
#define OFF_G2    54272    // f32 [16][12][12] = 9216    (gaze11 out)
#define OFF_A3    63488    // f32 [16][81] = 5184        (conv3 out)
#define OFF_A4    68672    // f32 [784] = 3136           (conv4 out / flatten order)
#define OFF_H1    71808    // f32 [64] = 256             (fc1 out)
#define OFF_KG11  72064    // f32 [121] = 484 (pad 512)  (gaze11 kernel)
#define OFF_W3F   72576    // f32 [2304] = 9216
#define OFF_W4F   81792    // f32 [2304] = 9216          -> ends 91008 < 112896
// persistent:
#define OFF_A1    112896   // f16 [16][27][32] = 27648   (conv1 out, pad row, zeroed cols)
#define OFF_KGS   140544   // f16 [2][2][16][16] = 2048  (gaze26 B, swizzled)
#define OFF_W1S   142592   // f16 [7][2][16][16] = 7168  (conv1 B, swizzled)
#define LDS_BYTES 149760

__device__ __forceinline__ float leaky(float x) { return x > 0.f ? x : 0.01f * x; }

__device__ __forceinline__ v8f wmma16(v16h a, v16h b, v8f c) {
  return __builtin_amdgcn_wmma_f32_16x16x32_f16(false, a, false, b, (short)0, c, false, false);
}

// half-slot i of a wave32 16-bit A/B fragment holds K-position (within a 32-chunk):
__device__ __forceinline__ constexpr int kpos(int i, int g) {
  return (i < 8) ? (g * 8 + i) : (8 + g * 8 + i);
}
// conv2 K reorder: row rc = c*5+kh (padded to 8 kw slots); offset of (rc, kw=0) in g1 halves
__device__ __forceinline__ constexpr int koff2(int rc) {
  return (rc / 5) * 864 + (rc % 5) * 32;   // c*27*32 + kh*32
}

__global__ void zero_out_kernel(float* out) {
  int i = threadIdx.x;
  if (i < 192) out[i] = 0.f;
}

__global__ void __launch_bounds__(256)
trex_frame_kernel(const float* __restrict__ traj_i, const float* __restrict__ traj_j,
                  const float* __restrict__ g26i,   const float* __restrict__ g26j,
                  const float* __restrict__ g11i,   const float* __restrict__ g11j,
                  const float* __restrict__ w1,  const float* __restrict__ b1,
                  const float* __restrict__ w2,  const float* __restrict__ b2,
                  const float* __restrict__ w3,  const float* __restrict__ b3,
                  const float* __restrict__ w4,  const float* __restrict__ b4,
                  const float* __restrict__ fw1, const float* __restrict__ fb1,
                  const float* __restrict__ fw2, const float* __restrict__ fb2,
                  float* __restrict__ out)
{
  __shared__ __align__(16) unsigned char smem[LDS_BYTES];
  float*    s_in  = (float*)(smem);             // DMA target
  _Float16* g1h   = (_Float16*)(smem + OFF_G1);
  _Float16* a2h   = (_Float16*)(smem + OFF_A2);
  _Float16* w2s   = (_Float16*)(smem + OFF_W2S);
  float*    g2f   = (float*)(smem + OFF_G2);
  float*    a3f   = (float*)(smem + OFF_A3);
  float*    a4f   = (float*)(smem + OFF_A4);
  float*    h1f   = (float*)(smem + OFF_H1);
  float*    kg11f = (float*)(smem + OFF_KG11);
  float*    w3f   = (float*)(smem + OFF_W3F);
  float*    w4f   = (float*)(smem + OFF_W4F);
  _Float16* a1h   = (_Float16*)(smem + OFF_A1);
  _Float16* kgs   = (_Float16*)(smem + OFF_KGS);
  _Float16* w1s   = (_Float16*)(smem + OFF_W1S);

  const int tid  = threadIdx.x;
  const int lane = tid & 31;
  const int m    = lane & 15;
  const int grp  = lane >> 4;
  // wave id as a *scalar* (SGPR): wave-uniform guards become s_cmp/s_cbranch,
  // EXEC stays all-ones around every WMMA (ISA requirement).
  const int wave_s = __builtin_amdgcn_readfirstlane(tid >> 5);

  const int f    = blockIdx.x;   // t*64 + b
  const int pair = blockIdx.y;
  const int b    = f & 63;

  const float* traj = pair ? traj_j : traj_i;
  const float* g26  = pair ? g26j  : g26i;
  const float* g11  = pair ? g11j  : g11i;

  // ---- Phase 0: async DMA of the frame (112896 B contiguous) into LDS ----
  {
    const float* fbase = traj + (size_t)f * (84 * 84 * 4);
    uint32_t lbase = (uint32_t)(uintptr_t)s_in;
    for (int ch = tid; ch < 7056; ch += 256) {
      uint32_t loff = lbase + (uint32_t)ch * 16u;
      uint64_t ga   = (uint64_t)(uintptr_t)(fbase + ch * 4);
      asm volatile("global_load_async_to_lds_b128 %0, %1, off"
                   :: "v"(loff), "v"(ga) : "memory");
    }
  }

  // ---- Stage (in DMA shadow): conv1 B (swizzled), gaze26 B (swizzled), zero a1 ----
  __builtin_prefetch(fw1 + tid, 0, 0);
  // conv1 B: K'' = kh*32 + (kw*4 + c), zeros at s>=28
  for (int e = tid; e < 7 * 2 * 16 * 16; e += 256) {
    int i = e & 15, n = (e >> 4) & 15, g = (e >> 8) & 1, kh = e >> 9;
    int s = kpos(i, g);
    int c = s & 3, kw = s >> 2;
    w1s[e] = (_Float16)((s < 28) ? w1[n * 196 + c * 49 + kh * 7 + kw] : 0.f);
  }
  // gaze26 B: K' = ((g-0.3)/0.7)/0.3, zeros outside 26x26
  for (int e = tid; e < 2 * 2 * 16 * 16; e += 256) {
    int i = e & 15, n = (e >> 4) & 15, g = (e >> 8) & 1, nt = e >> 9;
    int kk = kpos(i, g), col = nt * 16 + n;
    float v = 0.f;
    if (kk < 26 && col < 26)
      v = ((g26[b * 676 + kk * 26 + col] - 0.3f) * (1.0f / 0.7f)) * (1.0f / 0.3f);
    kgs[e] = (_Float16)v;
  }
  // zero a1 (pad cols must be 0 so gaze K-padding is exact)
  {
    uint4 z = {0u, 0u, 0u, 0u};
    uint4* a1v = (uint4*)a1h;
    for (int e = tid; e < 27648 / 16; e += 256) a1v[e] = z;
  }

  asm volatile("s_wait_asynccnt 0x0" ::: "memory");
  __syncthreads();

  // ---- Conv1 (7x7 s3, 4->16): implicit GEMM [676,224]x[224,16], WMMA f16 ----
  {
    // hoist all 7 B fragments (56 VGPRs) once per wave
    union { v16h v; uint4 q[2]; } Bf[7];
    #pragma unroll
    for (int ks = 0; ks < 7; ks++) {
      const uint4* wp = (const uint4*)(w1s + ((ks * 2 + grp) * 16 + m) * 16);
      Bf[ks].q[0] = wp[0]; Bf[ks].q[1] = wp[1];
    }
    float bn = b1[m];
    #pragma unroll
    for (int t = 0; t < 6; t++) {
      int tile = wave_s + t * 8;
      if (tile < 43) {                       // one scalar guard per tile
        int p = tile * 16 + m;
        int pc = p < 675 ? p : 675;
        int oh = pc / 26, ow = pc - oh * 26;
        const float* base = s_in + (oh * 252 + ow * 3) * 4 + grp * 8;
        v8f acc = {0.f,0.f,0.f,0.f,0.f,0.f,0.f,0.f};
        #pragma unroll
        for (int ks = 0; ks < 7; ks++) {     // ks == kh; contiguous K runs
          const float* src = base + ks * 336;
          float4 q0 = *(const float4*)(src);
          float4 q1 = *(const float4*)(src + 4);
          float4 q2 = *(const float4*)(src + 16);
          float4 q3 = *(const float4*)(src + 20);
          v16h a;
          a[0]=(_Float16)q0.x;  a[1]=(_Float16)q0.y;  a[2]=(_Float16)q0.z;  a[3]=(_Float16)q0.w;
          a[4]=(_Float16)q1.x;  a[5]=(_Float16)q1.y;  a[6]=(_Float16)q1.z;  a[7]=(_Float16)q1.w;
          a[8]=(_Float16)q2.x;  a[9]=(_Float16)q2.y;  a[10]=(_Float16)q2.z; a[11]=(_Float16)q2.w;
          a[12]=(_Float16)q3.x; a[13]=(_Float16)q3.y; a[14]=(_Float16)q3.z; a[15]=(_Float16)q3.w;
          acc = wmma16(a, Bf[ks].v, acc);
        }
        #pragma unroll
        for (int r = 0; r < 8; r++) {
          int pd = tile * 16 + r + grp * 8;  // <= 687 -> pad row 26
          int hh = pd / 26, ww = pd - hh * 26;
          a1h[m * 864 + hh * 32 + ww] = (_Float16)leaky(acc[r] + bn);
        }
      }
    }
  }
  __syncthreads();

  // ---- Stage for later phases (region disjoint from gaze26 traffic) ----
  // conv2 B: K'' = (c*5+kh)*8 + kw, zeros at kw>=5
  for (int e = tid; e < 20 * 2 * 16 * 16; e += 256) {
    int i = e & 15, n = (e >> 4) & 15, g = (e >> 8) & 1, ks = e >> 9;
    int s = kpos(i, g);
    int rc = ks * 4 + (s >> 3), kw = s & 7;
    int c = rc / 5, kh = rc - c * 5;
    w2s[e] = (_Float16)((kw < 5) ? w2[(n * 16 + c) * 25 + kh * 5 + kw] : 0.f);
  }
  for (int e = tid; e < 121; e += 256)
    kg11f[e] = ((g11[b * 121 + e] - 0.3f) * (1.0f / 0.7f)) * (1.0f / 0.3f);
  for (int e = tid; e < 2304; e += 256) { w3f[e] = w3[e]; w4f[e] = w4[e]; }

  // ---- Gaze26: per channel [26,26] x K'[26,26], WMMA (K padded to 32) ----
  {
    union { v16h v; uint4 q[2]; } B0, B1;   // channel-invariant, load once
    const uint4* k0p = (const uint4*)(kgs + ((0 * 2 + grp) * 16 + m) * 16);
    const uint4* k1p = (const uint4*)(kgs + ((1 * 2 + grp) * 16 + m) * 16);
    B0.q[0] = k0p[0]; B0.q[1] = k0p[1];
    B1.q[0] = k1p[0]; B1.q[1] = k1p[1];
    #pragma unroll
    for (int ci = 0; ci < 2; ci++) {
      int c = wave_s + ci * 8;              // scalar channel id
      #pragma unroll
      for (int mt = 0; mt < 2; mt++) {
        int h = mt * 16 + m; h = h < 26 ? h : 26;   // pad row (finite garbage)
        union { v16h v; uint4 q[2]; } A;
        A.q[0] = *(const uint4*)(a1h + c * 864 + h * 32 + grp * 8);
        A.q[1] = *(const uint4*)(a1h + c * 864 + h * 32 + grp * 8 + 16);
        v8f d0 = {0.f,0.f,0.f,0.f,0.f,0.f,0.f,0.f};
        v8f d1 = {0.f,0.f,0.f,0.f,0.f,0.f,0.f,0.f};
        d0 = wmma16(A.v, B0.v, d0);
        d1 = wmma16(A.v, B1.v, d1);
        #pragma unroll
        for (int r = 0; r < 8; r++) {
          int hh = mt * 16 + r + grp * 8;
          int row = hh < 26 ? hh : 26;      // clamp into pad row
          g1h[c * 864 + row * 32 + m]      = (_Float16)d0[r];
          g1h[c * 864 + row * 32 + 16 + m] = (_Float16)d1[r];  // cols 26..31 -> pad
        }
      }
    }
  }
  __syncthreads();

  // ---- Conv2 (5x5 s2, 16->16): implicit GEMM [121,640]x[640,16], WMMA f16 ----
  {
    int tile = wave_s;                     // 8 tiles cover M=121
    int p = tile * 16 + m;
    int pc = p < 120 ? p : 120;
    int oh = pc / 11, ow = pc - oh * 11;
    int baseA = oh * 64 + ow * 2;          // halves offset inside a channel plane
    v8f acc = {0.f,0.f,0.f,0.f,0.f,0.f,0.f,0.f};
    #pragma unroll
    for (int ks = 0; ks < 20; ks++) {
      union { v16h v; uint4 q[2]; } B;
      const uint4* wp = (const uint4*)(w2s + ((ks * 2 + grp) * 16 + m) * 16);
      B.q[0] = wp[0]; B.q[1] = wp[1];
      int off0 = grp ? koff2(ks * 4 + 1) : koff2(ks * 4 + 0);
      int off1 = grp ? koff2(ks * 4 + 3) : koff2(ks * 4 + 2);
      const _Float16* p0 = g1h + off0 + baseA;
      const _Float16* p1 = g1h + off1 + baseA;
      union { v16h v; unsigned int u[8]; } A;
      A.u[0] = *(const unsigned int*)(p0);     A.u[1] = *(const unsigned int*)(p0 + 2);
      A.u[2] = *(const unsigned int*)(p0 + 4); A.u[3] = *(const unsigned int*)(p0 + 6);
      A.u[4] = *(const unsigned int*)(p1);     A.u[5] = *(const unsigned int*)(p1 + 2);
      A.u[6] = *(const unsigned int*)(p1 + 4); A.u[7] = *(const unsigned int*)(p1 + 6);
      acc = wmma16(A.v, B.v, acc);
    }
    float bn = b2[m];
    #pragma unroll
    for (int r = 0; r < 8; r++) {
      int pd = tile * 16 + r + grp * 8;    // <=127 -> pad row 11
      int hh = pd / 11, ww = pd - hh * 11;
      a2h[m * 192 + hh * 16 + ww] = (_Float16)leaky(acc[r] + bn);
    }
  }
  __syncthreads();

  // ---- Gaze11 (16 x 11x11, K=11) on VALU, all-LDS ----
  {
    int c = tid & 15;
    for (int pos = tid >> 4; pos < 121; pos += 16) {
      int h = pos / 11, k = pos - h * 11;
      float s = 0.f;
      #pragma unroll
      for (int w = 0; w < 11; w++)
        s += (float)a2h[c * 192 + h * 16 + w] * kg11f[w * 11 + k];
      g2f[(c * 12 + h) * 12 + k] = s;
    }
  }
  __syncthreads();

  // ---- Conv3 (3x3 s1, 16->16) on VALU, all-LDS ----
  {
    int oc = tid & 15;
    for (int pos = tid >> 4; pos < 81; pos += 16) {
      int y = pos / 9, x = pos - y * 9;
      float s = b3[oc];
      for (int c = 0; c < 16; c++)
        #pragma unroll
        for (int kh = 0; kh < 3; kh++)
          #pragma unroll
          for (int kw = 0; kw < 3; kw++)
            s += g2f[(c * 12 + y + kh) * 12 + x + kw] * w3f[((oc * 16 + c) * 3 + kh) * 3 + kw];
      a3f[oc * 81 + pos] = leaky(s);
    }
  }
  __syncthreads();

  // ---- Conv4 (3x3 s1, 16->16) on VALU; flatten order c*49+y*7+x ----
  {
    int oc = tid & 15;
    for (int pos = tid >> 4; pos < 49; pos += 16) {
      int y = pos / 7, x = pos - y * 7;
      float s = b4[oc];
      for (int c = 0; c < 16; c++)
        #pragma unroll
        for (int kh = 0; kh < 3; kh++)
          #pragma unroll
          for (int kw = 0; kw < 3; kw++)
            s += a3f[c * 81 + (y + kh) * 9 + (x + kw)] * w4f[((oc * 16 + c) * 3 + kh) * 3 + kw];
      a4f[oc * 49 + pos] = leaky(s);
    }
  }
  __syncthreads();

  // ---- FC1: 784 -> 64 (float4 loads, fw1 is L2-resident) ----
  if (tid < 64) {
    const float4* wr = (const float4*)(fw1 + tid * 784);
    const float4* av = (const float4*)a4f;
    float s = fb1[tid];
    for (int i = 0; i < 196; i++) {
      float4 wv = wr[i], xv = av[i];
      s += wv.x * xv.x + wv.y * xv.y + wv.z * xv.z + wv.w * xv.w;
    }
    h1f[tid] = leaky(s);
  }
  __syncthreads();

  // ---- FC2: 64 -> 1; accumulate per-frame reward ----
  if (tid == 0) {
    float r = fb2[0];
    for (int j = 0; j < 64; j++) r += h1f[j] * fw2[j];
    atomicAdd(&out[pair * 64 + b], r);
    atomicAdd(&out[128 + b], fabsf(r));
  }
}

extern "C" void kernel_launch(void* const* d_in, const int* in_sizes, int n_in,
                              void* d_out, int out_size, void* d_ws, size_t ws_size,
                              hipStream_t stream) {
  (void)in_sizes; (void)n_in; (void)out_size; (void)d_ws; (void)ws_size;
  const float* traj_i = (const float*)d_in[0];
  const float* traj_j = (const float*)d_in[1];
  const float* g26i   = (const float*)d_in[2];
  const float* g26j   = (const float*)d_in[3];
  const float* g11i   = (const float*)d_in[4];
  const float* g11j   = (const float*)d_in[5];
  const float* w1  = (const float*)d_in[6];
  const float* b1  = (const float*)d_in[7];
  const float* w2  = (const float*)d_in[8];
  const float* b2  = (const float*)d_in[9];
  const float* w3  = (const float*)d_in[10];
  const float* b3  = (const float*)d_in[11];
  const float* w4  = (const float*)d_in[12];
  const float* b4  = (const float*)d_in[13];
  const float* fw1 = (const float*)d_in[14];
  const float* fb1 = (const float*)d_in[15];
  const float* fw2 = (const float*)d_in[16];
  const float* fb2 = (const float*)d_in[17];
  float* out = (float*)d_out;

  zero_out_kernel<<<1, 256, 0, stream>>>(out);
  dim3 grid(1600, 2);
  trex_frame_kernel<<<grid, 256, 0, stream>>>(traj_i, traj_j, g26i, g26j, g11i, g11j,
                                              w1, b1, w2, b2, w3, b3, w4, b4,
                                              fw1, fb1, fw2, fb2, out);
}
